// GAT_plm_dssp_edgefeat_sagpool_76390288327218
// MI455X (gfx1250) — matrix operand order repaired and checked by reading, not compiled, not run on hardware
//
#include <hip/hip_runtime.h>
#include <hip/hip_bf16.h>
#include <stdint.h>

namespace {
constexpr int B_     = 64;
constexpr int NPG_   = 512;
constexpr int EPG_   = 8192;
constexpr int FIN    = 1038;
constexpr int H_     = 128;
constexpr int NN     = B_ * NPG_;    // 32768
constexpr int EE     = B_ * EPG_;    // 524288
constexpr int EPN    = EE + NN;      // 557056  (edges + self loops)
constexpr int KEEP   = 103;          // ceil(0.2 * 512)
constexpr int KTILES = (FIN + 31) / 32;  // 33 (K padded to 1056)
constexpr float EPS_   = 1e-5f;
constexpr float SLOPE  = 0.01f;      // module LeakyReLU
constexpr float GSLOPE = 0.2f;       // GAT attention LeakyReLU
}

typedef __bf16 bf16_t;
typedef __attribute__((ext_vector_type(2)))  __bf16 v2bf;
typedef __attribute__((ext_vector_type(16))) __bf16 v16bf;
typedef __attribute__((ext_vector_type(8)))  float  v8f;

// packed fp32 -> bf16 (RNE): builtin if present, else native instruction via asm
__device__ __forceinline__ v2bf pk2(float a, float b) {
#if __has_builtin(__builtin_amdgcn_cvt_pk_bf16_f32)
  return __builtin_amdgcn_cvt_pk_bf16_f32(a, b);
#else
  unsigned r;
  asm("v_cvt_pk_bf16_f32 %0, %1, %2" : "=v"(r) : "v"(a), "v"(b));
  return __builtin_bit_cast(v2bf, r);
#endif
}
__device__ __forceinline__ bf16_t f2bf(float f) { return pk2(f, f)[0]; }
__device__ __forceinline__ float lrelu(float x, float s) { return x > 0.f ? x : s * x; }
// monotone float->int key for signed-int atomicMax
__device__ __forceinline__ int fkey(float f) {
  int b = __float_as_int(f);
  return b >= 0 ? b : (b ^ 0x7FFFFFFF);
}
__device__ __forceinline__ float funkey(int k) {
  int b = k >= 0 ? k : (k ^ 0x7FFFFFFF);
  return __int_as_float(b);
}

// ---------------------------------------------------------------- utilities
__global__ void fill_f32(float* p, float v, int n) {
  int i = blockIdx.x * blockDim.x + threadIdx.x;
  if (i < n) p[i] = v;
}
__global__ void fill_i32(int* p, int v, int n) {
  int i = blockIdx.x * blockDim.x + threadIdx.x;
  if (i < n) p[i] = v;
}

// ------------------------------------------------- pack W_gat -> bf16 B-layout
// Wb[((kt*8+nt)*32+lane)*16+j] holds W[k, col] with k = kt*32 + 16*(j>>3) + 8*(lane>>4) + (j&7),
// col = nt*16 + (lane&15); zero-padded past K=1038.  One contiguous 32B chunk per
// lane per k-tile in the GEMM.
__global__ void pack_w(const float* __restrict__ W, bf16_t* __restrict__ Wb) {
  int i = blockIdx.x * blockDim.x + threadIdx.x;
  if (i >= KTILES * 8 * 32 * 16) return;
  int j    = i & 15;
  int lane = (i >> 4) & 31;
  int nt   = (i >> 9) & 7;
  int kt   = i >> 12;
  int k    = kt * 32 + ((j >> 3) << 4) + ((lane >> 4) << 3) + (j & 7);
  int col  = nt * 16 + (lane & 15);
  float v  = (k < FIN) ? W[(size_t)k * H_ + col] : 0.f;
  Wb[i] = f2bf(v);
}

// -------------------------------------------------------- main GEMM: xp = x @ W
// grid (NN/16, 2 branches), block 256 = 8 waves; wave w computes the 16x16 tile
// covering output columns [16w, 16w+16).  fp32 x is converted to bf16 with
// v_cvt_pk_bf16_f32 and fed to v_wmma_f32_16x16x32_bf16 (fp32 accumulate).
// K tail (kt=32) is peeled so the 32 hot iterations are branch-free.
__global__ __launch_bounds__(256) void gemm_xp(const float* __restrict__ x1,
                                               const float* __restrict__ x2,
                                               const bf16_t* __restrict__ Wb,
                                               float* __restrict__ xp) {
  const int branch = blockIdx.y;
  const float* x = branch ? x2 : x1;
  float* out = xp + (size_t)branch * NN * H_;
  const int wave = threadIdx.x >> 5;
  const int lane = threadIdx.x & 31;
  const int half = lane >> 4;
  const int m    = (blockIdx.x << 4) + (lane & 15);
  const float* xrow = x + (size_t)m * FIN;
  const bf16_t* wbp = Wb + ((size_t)wave * 32 + lane) * 16;
  v8f acc = {};
  for (int kt = 0; kt < KTILES - 1; ++kt) {
    // A layout: lane (m, half) needs K = [k0+8h, k0+8h+8) and [k0+16+8h, k0+24+8h)
    const int kA = kt * 32 + 8 * half;
    __builtin_prefetch(xrow + kA + 32, 0, 0);
    const float2* p0 = (const float2*)(xrow + kA);        // 8B aligned
    const float2* p1 = (const float2*)(xrow + kA + 16);
    float2 q0 = p0[0], q1 = p0[1], q2 = p0[2], q3 = p0[3];
    float2 r0 = p1[0], r1 = p1[1], r2 = p1[2], r3 = p1[3];
    v16bf a;
    v2bf t;
    t = pk2(q0.x, q0.y); a[0]  = t[0]; a[1]  = t[1];
    t = pk2(q1.x, q1.y); a[2]  = t[0]; a[3]  = t[1];
    t = pk2(q2.x, q2.y); a[4]  = t[0]; a[5]  = t[1];
    t = pk2(q3.x, q3.y); a[6]  = t[0]; a[7]  = t[1];
    t = pk2(r0.x, r0.y); a[8]  = t[0]; a[9]  = t[1];
    t = pk2(r1.x, r1.y); a[10] = t[0]; a[11] = t[1];
    t = pk2(r2.x, r2.y); a[12] = t[0]; a[13] = t[1];
    t = pk2(r3.x, r3.y); a[14] = t[0]; a[15] = t[1];
    const v16bf b = *(const v16bf*)(wbp + (size_t)kt * (8 * 32 * 16));
    acc = __builtin_amdgcn_wmma_f32_16x16x32_bf16(false, a, false, b,
                                                  (short)0, acc, false, false);
  }
  {  // peeled K tail: kt = 32, valid K 1024..1037, rest zero
    const int k0 = (KTILES - 1) * 32;
    v16bf a;
    for (int j = 0; j < 16; ++j) {
      int k = k0 + ((j >> 3) << 4) + (j & 7) + 8 * half;
      a[j] = f2bf(k < FIN ? xrow[k] : 0.f);
    }
    const v16bf b = *(const v16bf*)(wbp + (size_t)(KTILES - 1) * (8 * 32 * 16));
    acc = __builtin_amdgcn_wmma_f32_16x16x32_bf16(false, a, false, b,
                                                  (short)0, acc, false, false);
  }
  // D layout: element r -> row rbase+r, col ncol
  const int ncol  = (wave << 4) + (lane & 15);
  const int rbase = (blockIdx.x << 4) + (half << 3);
  for (int r = 0; r < 8; ++r)
    out[(size_t)(rbase + r) * H_ + ncol] = acc[r];
}

// ------------------------------------------- alpha_s / alpha_d = xp @ a_src/dst
__global__ void alpha_kernel(const float* __restrict__ xp, const float* __restrict__ asrc,
                             const float* __restrict__ adst, float* as_, float* ad_) {
  int wave = threadIdx.x >> 5, lane = threadIdx.x & 31;
  int row = blockIdx.x * (blockDim.x >> 5) + wave;
  if (row >= 2 * NN) return;
  const float* xr = xp + (size_t)row * H_;
  float s = 0.f, d = 0.f;
  for (int k = 0; k < 4; ++k) {
    int c = lane + 32 * k;
    float v = xr[c];
    s += v * asrc[c]; d += v * adst[c];
  }
  for (int off = 16; off; off >>= 1) { s += __shfl_down(s, off); d += __shfl_down(d, off); }
  if (lane == 0) { as_[row] = s; ad_[row] = d; }
}

// --------------------------------------------- edge softmax pass 1: e + seg-max
__global__ void edge_e(const int* ei1, const int* ei2, const float* as_,
                       const float* ad_, float* earr, int* mkey) {
  int br = blockIdx.y;
  int i = blockIdx.x * blockDim.x + threadIdx.x;
  if (i >= EPN) return;
  const int* ei = br ? ei2 : ei1;
  int s, d;
  if (i < EE) { s = ei[i]; d = ei[EE + i]; } else { s = d = i - EE; }  // self loop
  float e = lrelu(as_[br * NN + s] + ad_[br * NN + d], GSLOPE);
  earr[(size_t)br * EPN + i] = e;
  atomicMax(&mkey[br * NN + d], fkey(e));
}

// ------------------------------------------ pass 2: exp(e - max), seg-sum denom
__global__ void edge_exp(const int* ei1, const int* ei2, float* earr,
                         const int* mkey, float* denom) {
  int br = blockIdx.y;
  int i = blockIdx.x * blockDim.x + threadIdx.x;
  if (i >= EPN) return;
  const int* ei = br ? ei2 : ei1;
  int d = (i < EE) ? ei[EE + i] : (i - EE);
  size_t o = (size_t)br * EPN + i;
  float ex = __expf(earr[o] - funkey(mkey[br * NN + d]));
  earr[o] = ex;                       // overwrite e with exp
  atomicAdd(&denom[br * NN + d], ex);
}

// --------------------------- pass 3: out[dst] += (ex/denom[dst]) * xp[src]
__global__ void edge_agg(const int* ei1, const int* ei2, const float* earr,
                         const float* denom, const float* xp, float* gout) {
  int br = blockIdx.y;
  int wave = threadIdx.x >> 5, lane = threadIdx.x & 31;
  int i = blockIdx.x * (blockDim.x >> 5) + wave;
  if (i >= EPN) return;
  const int* ei = br ? ei2 : ei1;
  int s, d;
  if (i < EE) { s = ei[i]; d = ei[EE + i]; } else { s = d = i - EE; }
  float w = earr[(size_t)br * EPN + i] / denom[br * NN + d];
  const float4* xs = (const float4*)(xp + ((size_t)br * NN + s) * H_);
  float* od = gout + ((size_t)br * NN + d) * H_;
  float4 v = xs[lane];
  atomicAdd(od + lane * 4 + 0, w * v.x);
  atomicAdd(od + lane * 4 + 1, w * v.y);
  atomicAdd(od + lane * 4 + 2, w * v.z);
  atomicAdd(od + lane * 4 + 3, w * v.w);
}

// ------------------------------------------------------ + b_gat, LeakyReLU 0.01
__global__ void bias_leaky(float* h, const float* __restrict__ bgat) {
  int i = blockIdx.x * blockDim.x + threadIdx.x;
  if (i >= 2 * NN * H_) return;
  h[i] = lrelu(h[i] + bgat[i & (H_ - 1)], SLOPE);
}

// ----------------------------------------------- BatchNorm train-mode statistics
__global__ void bn_stats(const float* __restrict__ h, float* bnsum) {
  int br = blockIdx.y, col = threadIdx.x;
  int r0 = blockIdx.x * 128;
  float s = 0.f, s2 = 0.f;
  for (int r = 0; r < 128; ++r) {
    float v = h[((size_t)br * NN + r0 + r) * H_ + col];
    s += v; s2 += v * v;
  }
  atomicAdd(&bnsum[br * H_ + col], s);
  atomicAdd(&bnsum[256 + br * H_ + col], s2);
}

__global__ void bn_apply(float* h, const float* __restrict__ bnsum,
                         const float* __restrict__ gamma, const float* __restrict__ beta) {
  int i = blockIdx.x * blockDim.x + threadIdx.x;
  if (i >= 2 * NN * H_) return;
  int c  = i & (H_ - 1);
  int br = i / (NN * H_);
  float mu = bnsum[br * H_ + c] * (1.f / NN);
  float va = bnsum[256 + br * H_ + c] * (1.f / NN) - mu * mu;
  h[i] = (h[i] - mu) * rsqrtf(va + EPS_) * gamma[c] + beta[c];
}

// --------------------------------------- SAGPool GraphConv: agg = seg_sum(h[src])
__global__ void sag_agg(const int* ei1, const int* ei2, const float* h, float* agg) {
  int br = blockIdx.y;
  int wave = threadIdx.x >> 5, lane = threadIdx.x & 31;
  int i = blockIdx.x * (blockDim.x >> 5) + wave;
  if (i >= EE) return;
  const int* ei = br ? ei2 : ei1;
  int s = ei[i], d = ei[EE + i];
  const float4* xs = (const float4*)(h + ((size_t)br * NN + s) * H_);
  float* od = agg + ((size_t)br * NN + d) * H_;
  float4 v = xs[lane];
  atomicAdd(od + lane * 4 + 0, v.x);
  atomicAdd(od + lane * 4 + 1, v.y);
  atomicAdd(od + lane * 4 + 2, v.z);
  atomicAdd(od + lane * 4 + 3, v.w);
}

// -------------------------- score = tanh(agg@W_rel + h@W_root + b_sag), per node
__global__ void score_kernel(const float* __restrict__ agg, const float* __restrict__ h,
                             const float* __restrict__ Wrel, const float* __restrict__ Wroot,
                             const float* __restrict__ bsag, float* s) {
  int wave = threadIdx.x >> 5, lane = threadIdx.x & 31;
  int row = blockIdx.x * (blockDim.x >> 5) + wave;
  if (row >= 2 * NN) return;
  const float* ar = agg + (size_t)row * H_;
  const float* hr = h + (size_t)row * H_;
  float acc = 0.f;
  for (int k = 0; k < 4; ++k) {
    int c = lane + 32 * k;
    acc += ar[c] * Wrel[c] + hr[c] * Wroot[c];
  }
  for (int off = 16; off; off >>= 1) acc += __shfl_down(acc, off);
  if (lane == 0) s[row] = tanhf(acc + bsag[0]);
}

// ------------- per-graph top-K (bitonic sort, value desc / idx asc) + weighted mean
__global__ void topk_pool(const float* __restrict__ s, const float* __restrict__ h,
                          float* pooled) {
  int g = blockIdx.x, br = blockIdx.y, t = threadIdx.x;
  __shared__ float sv[NPG_];
  __shared__ int   si[NPG_];
  sv[t] = s[br * NN + g * NPG_ + t];
  si[t] = t;
  __syncthreads();
  for (int k = 2; k <= NPG_; k <<= 1)
    for (int j = k >> 1; j > 0; j >>= 1) {
      int ixj = t ^ j;
      if (ixj > t) {
        float va = sv[t], vb = sv[ixj];
        int ia = si[t], ib = si[ixj];
        bool abb  = (va > vb) || (va == vb && ia < ib);  // a belongs before b (desc)
        bool desc = ((t & k) == 0);
        if (desc != abb) { sv[t] = vb; sv[ixj] = va; si[t] = ib; si[ixj] = ia; }
      }
      __syncthreads();
    }
  if (t < H_) {
    float acc = 0.f;
    for (int j = 0; j < KEEP; ++j)
      acc += sv[j] * h[((size_t)br * NN + g * NPG_ + si[j]) * H_ + t];
    pooled[((size_t)br * B_ + g) * H_ + t] = acc * (1.f / KEEP);
  }
}

// ---------------------------------------------------------------- head MLP [B,1]
__global__ void head_mlp(const float* __restrict__ pooled, const float* __restrict__ W1,
                         const float* __restrict__ b1, const float* __restrict__ W2,
                         const float* __restrict__ b2, float* out) {
  int b = blockIdx.x, t = threadIdx.x;  // 128 threads
  const float* g1 = pooled + (size_t)b * H_;
  const float* g2 = pooled + (size_t)(B_ + b) * H_;
  float acc = b1[t];
  for (int c = 0; c < H_; ++c) acc += g1[c] * W1[c * H_ + t];
  for (int c = 0; c < H_; ++c) acc += g2[c] * W1[(H_ + c) * H_ + t];
  acc = lrelu(acc, SLOPE);
  __shared__ float red[H_];
  red[t] = acc * W2[t];
  __syncthreads();
  for (int off = 64; off; off >>= 1) {
    if (t < off) red[t] += red[t + off];
    __syncthreads();
  }
  if (t == 0) out[b] = 1.f / (1.f + __expf(-(red[0] + b2[0])));
}

extern "C" void kernel_launch(void* const* d_in, const int* in_sizes, int n_in,
                              void* d_out, int out_size, void* d_ws, size_t ws_size,
                              hipStream_t stream) {
  (void)in_sizes; (void)n_in; (void)out_size; (void)ws_size;
  const float* p1x  = (const float*)d_in[0];
  const int*   e1   = (const int*)d_in[1];
  const float* p2x  = (const float*)d_in[2];
  const int*   e2   = (const int*)d_in[3];
  const float* Wg   = (const float*)d_in[4];
  const float* asr  = (const float*)d_in[5];
  const float* ads  = (const float*)d_in[6];
  const float* bg   = (const float*)d_in[7];
  const float* gam  = (const float*)d_in[8];
  const float* bet  = (const float*)d_in[9];
  const float* Wrel = (const float*)d_in[10];
  const float* Wroot= (const float*)d_in[11];
  const float* bs   = (const float*)d_in[12];
  const float* W1   = (const float*)d_in[13];
  const float* b1   = (const float*)d_in[14];
  const float* W2   = (const float*)d_in[15];
  const float* b2   = (const float*)d_in[16];
  float* out = (float*)d_out;

  char* ws = (char*)d_ws;
  size_t off = 0;
  auto alloc = [&](size_t bytes) {
    size_t o = off;
    off = (off + bytes + 255) & ~(size_t)255;
    return o;
  };
  float*  xp     = (float*)(ws + alloc(2ull * NN * H_ * 4));   // xp, later reused as agg
  float*  hbuf   = (float*)(ws + alloc(2ull * NN * H_ * 4));   // GAT out -> h (in place)
  bf16_t* Wb     = (bf16_t*)(ws + alloc((size_t)KTILES * 8 * 32 * 16 * 2));
  float*  as_    = (float*)(ws + alloc(2ull * NN * 4));
  float*  ad_    = (float*)(ws + alloc(2ull * NN * 4));
  float*  earr   = (float*)(ws + alloc(2ull * EPN * 4));       // e, then exp(e-max)
  int*    mkey   = (int*)(ws + alloc(2ull * NN * 4));
  float*  denom  = (float*)(ws + alloc(2ull * NN * 4));
  float*  bnsum  = (float*)(ws + alloc(512ull * 4));           // [0,256): sum  [256,512): sumsq
  float*  sbuf   = (float*)(ws + alloc(2ull * NN * 4));
  float*  pooled = (float*)(ws + alloc(2ull * B_ * H_ * 4));

  // pack W into WMMA B-operand bf16 layout (L2-resident, reused by every tile)
  { int n = KTILES * 8 * 32 * 16; pack_w<<<(n + 255) / 256, 256, 0, stream>>>(Wg, Wb); }
  // init accumulators
  { int n = 2 * NN * H_; fill_f32<<<(n + 255) / 256, 256, 0, stream>>>(hbuf, 0.f, n); }
  { int n = 2 * NN; fill_i32<<<(n + 255) / 256, 256, 0, stream>>>(mkey, (int)0x80000000, n); }
  { int n = 2 * NN; fill_f32<<<(n + 255) / 256, 256, 0, stream>>>(denom, 0.f, n); }
  fill_f32<<<2, 256, 0, stream>>>(bnsum, 0.f, 512);

  // main HBM-bound WMMA GEMM (both branches via grid.y)
  gemm_xp<<<dim3(NN / 16, 2), 256, 0, stream>>>(p1x, p2x, Wb, xp);

  // attention logits and edge softmax
  alpha_kernel<<<(2 * NN) / 8, 256, 0, stream>>>(xp, asr, ads, as_, ad_);
  edge_e  <<<dim3((EPN + 255) / 256, 2), 256, 0, stream>>>(e1, e2, as_, ad_, earr, mkey);
  edge_exp<<<dim3((EPN + 255) / 256, 2), 256, 0, stream>>>(e1, e2, earr, mkey, denom);
  edge_agg<<<dim3(EPN / 8, 2), 256, 0, stream>>>(e1, e2, earr, denom, xp, hbuf);

  // bias + LeakyReLU + BatchNorm (train-mode stats), in place on hbuf
  { int n = 2 * NN * H_; bias_leaky<<<(n + 255) / 256, 256, 0, stream>>>(hbuf, bg); }
  bn_stats<<<dim3(NN / 128, 2), 128, 0, stream>>>(hbuf, bnsum);
  { int n = 2 * NN * H_; bn_apply<<<(n + 255) / 256, 256, 0, stream>>>(hbuf, bnsum, gam, bet); }

  // SAGPool scoring (xp buffer is free now -> reuse as agg)
  { int n = 2 * NN * H_; fill_f32<<<(n + 255) / 256, 256, 0, stream>>>(xp, 0.f, n); }
  sag_agg<<<dim3(EE / 8, 2), 256, 0, stream>>>(e1, e2, hbuf, xp);
  score_kernel<<<(2 * NN) / 8, 256, 0, stream>>>(xp, hbuf, Wrel, Wroot, bs, sbuf);

  // per-graph top-K + weighted mean pool, then head MLP
  topk_pool<<<dim3(B_, 2), NPG_, 0, stream>>>(sbuf, hbuf, pooled);
  head_mlp<<<B_, H_, 0, stream>>>(pooled, W1, b1, W2, b2, out);
}
